// LinearAttention_4767413699311
// MI455X (gfx1250) — compile-verified, matrix operand order checked by reading, not run on hardware
//
#include <hip/hip_runtime.h>
#include <hip/hip_fp16.h>

// ---------------------------------------------------------------------------
// Linear attention (HEADS=4, DIM_HEAD=32, C=128) for x:[32,64,64,128] fp32.
// All matrix math on v_wmma_f32_16x16x32_f16 (f16 in, f32 accumulate).
//
// Data flow (all intermediates ws-resident; ~97 MiB, fits MI455X 192 MB L2):
//   qkv_gemm : x @ w_qkv, x read ONCE (A-frags held in VGPRs, 24 col tiles
//              per wave).  q stored token-major [tok][128]; k,v stored
//              channel-major [b][ch][4096] via free WMMA-D transpose
//              (8 accumulator rows == 8 consecutive tokens -> one v8h store).
//   kstats   : k softmax stats over tokens, fully coalesced on k_t rows.
//   context  : k^T v per (b,h); fragments are 16B vector loads (K-pattern
//              n-runs of 8 are contiguous in channel-major layout).
//   final    : q softmax (lane-pair shfl_xor) -> q@context -> w_out GEMM
//              (LDS re-layout, ds_load_b128 A-frags) -> bias -> LayerNorm.
// ---------------------------------------------------------------------------

typedef __attribute__((ext_vector_type(16))) _Float16 v16h;
typedef __attribute__((ext_vector_type(8)))  _Float16 v8h;
typedef __attribute__((ext_vector_type(8)))  float    v8f;
typedef __attribute__((ext_vector_type(4)))  float    v4f;

#define NBATCH   32
#define NTOK_IMG 4096          // 64*64 tokens per image
#define NTOK     131072        // 32*4096
#define CIN      128
#define QKVC     384
#define NHEADS   4
#define DHEAD    32

// workspace layout (bytes)
#define WQKV_PACK_OFF 0u            // 24*4*32*16 half = 98304 B
#define WOUT_PACK_OFF 98304u        //  8*4*32*16 half = 32768 B
#define KMAX_OFF      131072u       // 32*128 f32     = 16384 B
#define KSUM_OFF      147456u       // 32*128 f32     = 16384 B
#define CTX_OFF       163840u       // 32*4*32*32 f32 = 524288 B
#define QH_OFF        688128u       // 131072*128 f16 = 32 MiB  (token-major)
#define KT_OFF        34242560u     // 32*128*4096 f16 = 32 MiB (channel-major)
#define VT_OFF        67796992u     // 32*128*4096 f16 = 32 MiB (channel-major)

// K-index pattern of the 16-bit A/B WMMA fragment for a lane's `half` (=lane>>4):
// elements 0..7  -> K = 8*half + e ;  elements 8..15 -> K = 16 + 8*half + (e-8)
__device__ __forceinline__ int kpat(int half, int e) {
  return (e < 8) ? (8 * half + e) : (16 + 8 * half + (e - 8));
}

__device__ __forceinline__ v8f wmma_f16(v16h a, v16h b, v8f c) {
  return __builtin_amdgcn_wmma_f32_16x16x32_f16(false, a, false, b, (short)0, c,
                                                false, false);
}

// ---------------------------------------------------------------------------
// K0: repack w_qkv [128,384] and w_out [128,128] into B-fragment layout:
//     [n_tile][kb][lane][16 halves]; lane holds column n_tile*16+(lane&15).
//     Also zero the context accumulator.
// ---------------------------------------------------------------------------
__global__ void pack_weights(const float* __restrict__ wqkv,
                             const float* __restrict__ wout,
                             _Float16* __restrict__ wqkv_p,
                             _Float16* __restrict__ wout_p,
                             float* __restrict__ ctx) {
  const int lane = threadIdx.x;          // 0..31
  const int bid  = blockIdx.x;           // 0..127
  const int half = lane >> 4;
  const int lcol = lane & 15;
  if (bid < 96) {                        // w_qkv: 24 n-tiles * 4 k-blocks
    const int kb  = bid & 3;
    const int col = (bid >> 2) * 16 + lcol;
    _Float16* dst = wqkv_p + ((size_t)bid * 32 + lane) * 16;
#pragma unroll
    for (int e = 0; e < 16; ++e) {
      int k = kb * 32 + kpat(half, e);
      dst[e] = (_Float16)wqkv[(size_t)k * QKVC + col];
    }
  } else {                               // w_out: 8 n-tiles * 4 k-blocks
    const int i   = bid - 96;
    const int kb  = i & 3;
    const int col = (i >> 2) * 16 + lcol;
    _Float16* dst = wout_p + ((size_t)i * 32 + lane) * 16;
#pragma unroll
    for (int e = 0; e < 16; ++e) {
      int k = kb * 32 + kpat(half, e);
      dst[e] = (_Float16)wout[(size_t)k * CIN + col];
    }
  }
  const int tid = bid * 32 + lane;
  for (int i = tid; i < NBATCH * NHEADS * DHEAD * DHEAD; i += 128 * 32)
    ctx[i] = 0.0f;
}

// ---------------------------------------------------------------------------
// K1: qkv = x @ w_qkv.  One wave per 16-token tile; the 4 A-fragments
//     (K=128) stay in VGPRs while the wave walks all 24 column tiles
//     (96 WMMAs) -> x is read exactly once from HBM.
//     nt 0..7  -> q  token-major    [tok][128]
//     nt 8..15 -> k  channel-major  [b][ch][4096]  (v8h transpose store)
//     nt 16..23-> v  channel-major  [b][ch][4096]
// ---------------------------------------------------------------------------
__global__ void qkv_gemm(const float* __restrict__ x,
                         const _Float16* __restrict__ wp,
                         _Float16* __restrict__ qh,
                         _Float16* __restrict__ kt,
                         _Float16* __restrict__ vt) {
  const int lane = threadIdx.x;
  const int mt   = blockIdx.x;           // token tile (16 tokens)
  const int half = lane >> 4;
  const int lcol = lane & 15;
  const int row  = mt * 16 + lcol;
  const int b    = (mt * 16) >> 12;      // image index
  const int nb   = (mt * 16) & (NTOK_IMG - 1);  // token offset within image

  // A fragments for the whole K=128
  v16h afr[4];
#pragma unroll
  for (int kb = 0; kb < 4; ++kb) {
    const float* ap = x + (size_t)row * CIN + kb * 32 + 8 * half;
    v4f f0 = *(const v4f*)(ap);
    v4f f1 = *(const v4f*)(ap + 4);
    v4f f2 = *(const v4f*)(ap + 16);
    v4f f3 = *(const v4f*)(ap + 20);
#pragma unroll
    for (int i = 0; i < 4; ++i) {
      afr[kb][i]      = (_Float16)f0[i];
      afr[kb][4 + i]  = (_Float16)f1[i];
      afr[kb][8 + i]  = (_Float16)f2[i];
      afr[kb][12 + i] = (_Float16)f3[i];
    }
  }

  for (int nt = 0; nt < 24; ++nt) {
    v8f acc = {};
#pragma unroll
    for (int kb = 0; kb < 4; ++kb) {
      v16h bf = *(const v16h*)(wp + (((size_t)nt * 4 + kb) * 32 + lane) * 16);
      acc = wmma_f16(afr[kb], bf, acc);
    }
    if (nt < 8) {                        // q: token-major
      const int col = nt * 16 + lcol;
#pragma unroll
      for (int r = 0; r < 8; ++r)
        qh[(size_t)(mt * 16 + r + 8 * half) * CIN + col] = (_Float16)acc[r];
    } else {                             // k/v: channel-major transpose store
      _Float16* dst = (nt < 16) ? kt : vt;
      const int ch = ((nt & 7) * 16) + lcol;
      v8h pk;
#pragma unroll
      for (int r = 0; r < 8; ++r) pk[r] = (_Float16)acc[r];
      *(v8h*)(dst + ((size_t)b * CIN + ch) * NTOK_IMG + nb + 8 * half) = pk;
    }
  }
}

// ---------------------------------------------------------------------------
// K2: per-(b, head*d) max and sum(exp) over the 4096-token axis (k softmax).
//     Fully coalesced on channel-major k_t rows.
// ---------------------------------------------------------------------------
__global__ void kstats(const _Float16* __restrict__ kt,
                       float* __restrict__ kmax, float* __restrict__ ksum) {
  __shared__ float red[256];
  const int tid = threadIdx.x;
  const _Float16* rowp = kt + (size_t)blockIdx.x * NTOK_IMG;  // blockIdx = b*128+kd

  float m = -1e30f;
  for (int n = tid; n < NTOK_IMG; n += 256)
    m = fmaxf(m, (float)rowp[n]);
  red[tid] = m; __syncthreads();
  for (int s = 128; s > 0; s >>= 1) {
    if (tid < s) red[tid] = fmaxf(red[tid], red[tid + s]);
    __syncthreads();
  }
  m = red[0]; __syncthreads();

  float s = 0.0f;
  for (int n = tid; n < NTOK_IMG; n += 256)
    s += __expf((float)rowp[n] - m);
  red[tid] = s; __syncthreads();
  for (int st = 128; st > 0; st >>= 1) {
    if (tid < st) red[tid] += red[tid + st];
    __syncthreads();
  }
  if (tid == 0) { kmax[blockIdx.x] = m; ksum[blockIdx.x] = red[0]; }
}

// ---------------------------------------------------------------------------
// K3: context[b,h,d,e] = sum_n softmax_k[n,d] * v[n,e]/N.   2x2 WMMA tiles,
//     K (=tokens) split 16 ways across blocks, f32 atomics into ws.
//     Each fragment = two 16B v8h loads per lane (contiguous n-runs).
// ---------------------------------------------------------------------------
__global__ void context_gemm(const _Float16* __restrict__ kt,
                             const _Float16* __restrict__ vt,
                             const float* __restrict__ kmax,
                             const float* __restrict__ ksum,
                             float* __restrict__ ctx) {
  const int lane = threadIdx.x;
  const int b = blockIdx.x, h = blockIdx.y, ns = blockIdx.z;
  const int half = lane >> 4;
  const int lcol = lane & 15;

  const size_t rbase = ((size_t)b * CIN + h * DHEAD + lcol) * NTOK_IMG;
  const _Float16* kr0 = kt + rbase;                      // d = lcol
  const _Float16* kr1 = kr0 + (size_t)16 * NTOK_IMG;     // d = 16+lcol
  const _Float16* vr0 = vt + rbase;                      // e = lcol
  const _Float16* vr1 = vr0 + (size_t)16 * NTOK_IMG;     // e = 16+lcol

  const int sb = b * CIN + h * DHEAD;
  const float m0  = kmax[sb + lcol];
  const float is0 = 1.0f / ksum[sb + lcol];
  const float m1  = kmax[sb + 16 + lcol];
  const float is1 = 1.0f / ksum[sb + 16 + lcol];
  const float invN = 1.0f / (float)NTOK_IMG;

  v8f acc00 = {}, acc01 = {}, acc10 = {}, acc11 = {};
  for (int kb = 0; kb < 8; ++kb) {       // 8 * 32 = 256 tokens per block
    const int n0 = ns * 256 + kb * 32 + 8 * half;  // elements 0..7
    const int n1 = n0 + 16;                        // elements 8..15
    v8h ka = *(const v8h*)(kr0 + n0), kb_ = *(const v8h*)(kr0 + n1);
    v8h kc = *(const v8h*)(kr1 + n0), kd_ = *(const v8h*)(kr1 + n1);
    v8h va = *(const v8h*)(vr0 + n0), vb_ = *(const v8h*)(vr0 + n1);
    v8h vc = *(const v8h*)(vr1 + n0), vd_ = *(const v8h*)(vr1 + n1);
    v16h a0, a1, bv0, bv1;
#pragma unroll
    for (int e = 0; e < 8; ++e) {
      a0[e]      = (_Float16)(__expf((float)ka[e]  - m0) * is0);
      a0[8 + e]  = (_Float16)(__expf((float)kb_[e] - m0) * is0);
      a1[e]      = (_Float16)(__expf((float)kc[e]  - m1) * is1);
      a1[8 + e]  = (_Float16)(__expf((float)kd_[e] - m1) * is1);
      bv0[e]     = (_Float16)((float)va[e]  * invN);
      bv0[8 + e] = (_Float16)((float)vb_[e] * invN);
      bv1[e]     = (_Float16)((float)vc[e]  * invN);
      bv1[8 + e] = (_Float16)((float)vd_[e] * invN);
    }
    acc00 = wmma_f16(a0, bv0, acc00);
    acc01 = wmma_f16(a0, bv1, acc01);
    acc10 = wmma_f16(a1, bv0, acc10);
    acc11 = wmma_f16(a1, bv1, acc11);
  }
  float* cb = ctx + (size_t)(b * NHEADS + h) * DHEAD * DHEAD;
#pragma unroll
  for (int r = 0; r < 8; ++r) {
    int d0 = r + 8 * half, d1 = 16 + r + 8 * half;
    atomicAdd(cb + d0 * DHEAD + lcol,      acc00[r]);
    atomicAdd(cb + d0 * DHEAD + 16 + lcol, acc01[r]);
    atomicAdd(cb + d1 * DHEAD + lcol,      acc10[r]);
    atomicAdd(cb + d1 * DHEAD + 16 + lcol, acc11[r]);
  }
}

// ---------------------------------------------------------------------------
// K4: per 16-token tile: q-softmax(d) -> q @ context -> w_out GEMM + bias
//     -> LayerNorm -> out.  One wave per tile; LDS for the re-layout.
// ---------------------------------------------------------------------------
__global__ void final_fused(const _Float16* __restrict__ qh,
                            const float* __restrict__ ctx,
                            const _Float16* __restrict__ woutp,
                            const float* __restrict__ b_out,
                            const float* __restrict__ ln_scale,
                            float* __restrict__ out) {
  __shared__ float attn[16 * 128];
  __shared__ float lino[16 * 128];
  const int lane = threadIdx.x;
  const int half = lane >> 4;
  const int lcol = lane & 15;
  const size_t tokbase = (size_t)blockIdx.x * 16;
  const int b = (int)(tokbase >> 12);              // 4096 tokens per image
  const float qscale = 0.17677669529663687f;       // 1/sqrt(32)

  for (int h = 0; h < NHEADS; ++h) {
    // --- q softmax over d (32 values split across lane pair L, L^16) ---
    const _Float16* qrow = qh + (tokbase + lcol) * CIN + h * DHEAD;
    v8h q0 = *(const v8h*)(qrow + 8 * half);
    v8h q1 = *(const v8h*)(qrow + 16 + 8 * half);
    float qv[16];
    float mloc = -1e30f;
#pragma unroll
    for (int e = 0; e < 8; ++e) {
      qv[e]     = (float)q0[e];
      qv[8 + e] = (float)q1[e];
    }
#pragma unroll
    for (int e = 0; e < 16; ++e) mloc = fmaxf(mloc, qv[e]);
    float mm = fmaxf(mloc, __shfl_xor(mloc, 16));
    float sloc = 0.0f;
#pragma unroll
    for (int e = 0; e < 16; ++e) { qv[e] = __expf(qv[e] - mm); sloc += qv[e]; }
    float tot = sloc + __shfl_xor(sloc, 16);
    float sc  = qscale / tot;
    v16h a;
#pragma unroll
    for (int e = 0; e < 16; ++e) a[e] = (_Float16)(qv[e] * sc);

    // --- context B fragments (32x32 per head, 2 column tiles) ---
    const float* cb = ctx + (size_t)(b * NHEADS + h) * DHEAD * DHEAD;
    v16h bv0, bv1;
#pragma unroll
    for (int e = 0; e < 16; ++e) {
      int d = kpat(half, e);
      bv0[e] = (_Float16)cb[d * DHEAD + lcol];
      bv1[e] = (_Float16)cb[d * DHEAD + 16 + lcol];
    }
    v8f c0 = {}, c1 = {};
    c0 = wmma_f16(a, bv0, c0);
    c1 = wmma_f16(a, bv1, c1);
#pragma unroll
    for (int r = 0; r < 8; ++r) {
      int rw = r + 8 * half;
      attn[rw * 128 + h * DHEAD + lcol]      = c0[r];
      attn[rw * 128 + h * DHEAD + 16 + lcol] = c1[r];
    }
  }
  __syncthreads();

  // --- out = attn @ w_out + b_out ---
  for (int nt = 0; nt < 8; ++nt) {
    v8f acc = {};
#pragma unroll
    for (int kb = 0; kb < 4; ++kb) {
      const float* ar = &attn[lcol * 128 + kb * 32 + 8 * half];
      v4f f0 = *(const v4f*)(ar);
      v4f f1 = *(const v4f*)(ar + 4);
      v4f f2 = *(const v4f*)(ar + 16);
      v4f f3 = *(const v4f*)(ar + 20);
      v16h a;
#pragma unroll
      for (int i = 0; i < 4; ++i) {
        a[i]      = (_Float16)f0[i];
        a[4 + i]  = (_Float16)f1[i];
        a[8 + i]  = (_Float16)f2[i];
        a[12 + i] = (_Float16)f3[i];
      }
      v16h bf = *(const v16h*)(woutp + (((size_t)nt * 4 + kb) * 32 + lane) * 16);
      acc = wmma_f16(a, bf, acc);
    }
    const int c = nt * 16 + lcol;
    const float bias = b_out[c];
#pragma unroll
    for (int r = 0; r < 8; ++r)
      lino[(r + 8 * half) * 128 + c] = acc[r] + bias;
  }
  __syncthreads();

  // --- LayerNorm over 128 channels, one row per lane 0..15 ---
  if (lane < 16) {
    float mu = 0.0f;
    for (int c = 0; c < 128; ++c) mu += lino[lane * 128 + c];
    mu *= (1.0f / 128.0f);
    float var = 0.0f;
    for (int c = 0; c < 128; ++c) {
      float d = lino[lane * 128 + c] - mu;
      var += d * d;
    }
    var *= (1.0f / 128.0f);
    const float inv = rsqrtf(var + 1e-5f);
    const size_t o = (tokbase + lane) * CIN;
    for (int c = 0; c < 128; ++c)
      out[o + c] = (lino[lane * 128 + c] - mu) * inv * ln_scale[c];
  }
}

// ---------------------------------------------------------------------------
extern "C" void kernel_launch(void* const* d_in, const int* in_sizes, int n_in,
                              void* d_out, int out_size, void* d_ws, size_t ws_size,
                              hipStream_t stream) {
  (void)in_sizes; (void)n_in; (void)out_size; (void)ws_size;
  const float* x    = (const float*)d_in[0];
  const float* wqkv = (const float*)d_in[1];
  const float* wout = (const float*)d_in[2];
  const float* bo   = (const float*)d_in[3];
  const float* lns  = (const float*)d_in[4];
  float* out = (float*)d_out;

  char* ws = (char*)d_ws;
  _Float16* wqkv_p = (_Float16*)(ws + WQKV_PACK_OFF);
  _Float16* wout_p = (_Float16*)(ws + WOUT_PACK_OFF);
  float*    kmax   = (float*)(ws + KMAX_OFF);
  float*    ksum   = (float*)(ws + KSUM_OFF);
  float*    ctx    = (float*)(ws + CTX_OFF);
  _Float16* qh     = (_Float16*)(ws + QH_OFF);
  _Float16* kt     = (_Float16*)(ws + KT_OFF);
  _Float16* vt     = (_Float16*)(ws + VT_OFF);

  pack_weights<<<128, 32, 0, stream>>>(wqkv, wout, wqkv_p, wout_p, ctx);
  qkv_gemm<<<NTOK / 16, 32, 0, stream>>>(x, wqkv_p, qh, kt, vt);
  kstats<<<NBATCH * CIN, 256, 0, stream>>>(kt, kmax, ksum);
  context_gemm<<<dim3(NBATCH, NHEADS, 16), 32, 0, stream>>>(kt, vt, kmax, ksum, ctx);
  final_fused<<<NTOK / 16, 32, 0, stream>>>(qh, ctx, wout_p, bo, lns, out);
}
